// GameProcessor_59356448031219
// MI455X (gfx1250) — compile-verified, matrix operand order checked by reading, not run on hardware
//
#include <hip/hip_runtime.h>
#include <hip/hip_bf16.h>
#include <math.h>

// ---------------------------------------------------------------------------
// Persistent-cluster GRU rating scan for MI455X (gfx1250, wave32, WMMA).
// 16 workgroups keep the full (bf16) weight set resident in LDS across all
// 30000 sequential steps; per-step GEMVs run on v_wmma_f32_16x16x32_bf16.
// Only 2 grid barriers per step: the GRU/LayerNorm finale is computed
// redundantly by every WG, and the updated embedding rows are forwarded
// step-to-step through a local LDS double buffer (global writes by WG0 are
// only consumed >=2 barrier-pairs later).
// ---------------------------------------------------------------------------

typedef __attribute__((ext_vector_type(16))) __bf16 v16bf;
typedef __attribute__((ext_vector_type(8)))  __bf16 v8bf;
typedef __attribute__((ext_vector_type(8)))  float  v8f;

#define T_TEAMS 364
#define D_DIM   512
#define F_DIM   28
#define G_GAMES 30000
#define H_DIM   128
#define CIN_    1053
#define CINP    1056           // CIN padded to multiple of 32
#define KT_P    (CINP/32)      // 33 k-tiles for the projection GEMM
#define KT_D    (D_DIM/32)     // 16 k-tiles for D=512 GEMMs
#define NWG     16
#define WGSIZE  256
#define NWAVES  8

// dynamic-LDS fragment pools (bf16 elements), per workgroup
#define WP_ELEMS   (2*KT_P*512)   // 2 proj column-tiles owned per WG
#define WHH_ELEMS  (6*KT_D*512)   // 6 gh column-tiles
#define WIH_ELEMS  (6*KT_D*512)   // 6 gi column-tiles
#define WI1_ELEMS  (1*KT_D*512)   // 1 importance tile (WGs 0..7)
#define DYN_ELEMS  (WP_ELEMS + WHH_ELEMS + WIH_ELEMS + WI1_ELEMS)
#define DYN_BYTES  ((size_t)DYN_ELEMS * 2)

// fragment element -> K offset, per CDNA5 16-bit A/B VGPR striping
// (CDNA5 ISA §7.12.2): e 0..7 -> 8*kh+e ; e 8..15 -> 16+8*kh+(e-8)
__device__ __forceinline__ int frag_k(int e, int kh) {
    return (e < 8) ? (8 * kh + e) : (16 + 8 * kh + (e - 8));
}

// Convert a slice of a row-major (N,K) f32 weight matrix into bf16 WMMA
// B-fragments stored contiguously per lane in LDS (prologue only).
__device__ void load_frags(__bf16* dst, const float* W, int Kvalid,
                           int colBase, int ntiles, int KT, int tid) {
    const int lane = tid & 31, slot = tid >> 5;
    const int ncol = lane & 15, kh = lane >> 4;
    for (int fr = slot; fr < ntiles * KT; fr += NWAVES) {
        int t = fr / KT, kt = fr - t * KT;
        int n = colBase + t * 16 + ncol;                 // row of W = output col
        __bf16* p = dst + (size_t)((t * KT + kt) * 32 + lane) * 16;
#pragma unroll
        for (int e = 0; e < 16; ++e) {
            int k = kt * 32 + frag_k(e, kh);
            float v = (k < Kvalid) ? W[(size_t)n * Kvalid + k] : 0.0f;
            p[e] = (__bf16)v;
        }
    }
}

// A-fragment (rows 0,1 live; rows >=2 read a permanent zero row): two
// unconditional, 16B-aligned ds_load_b128s. No predication.
__device__ __forceinline__ v16bf build_afrag(const __bf16* buf, int stride,
                                             int k0, int lane) {
    const int m = lane & 15, kh = lane >> 4;
    const int row = (m < 2) ? m : 2;                 // row 2 is all zeros
    const __bf16* p = buf + row * stride + k0 + 8 * kh;
    v8bf lo = *(const v8bf*)(p);
    v8bf hi = *(const v8bf*)(p + 16);
    return __builtin_shufflevector(lo, hi, 0, 1, 2, 3, 4, 5, 6, 7,
                                   8, 9, 10, 11, 12, 13, 14, 15);
}

__device__ __forceinline__ void grid_barrier(int* cnt, int* gen) {
    __threadfence();
    __syncthreads();
    if (threadIdx.x == 0) {
        int g = __hip_atomic_load(gen, __ATOMIC_RELAXED, __HIP_MEMORY_SCOPE_AGENT);
        int arrived = __hip_atomic_fetch_add(cnt, 1, __ATOMIC_ACQ_REL,
                                             __HIP_MEMORY_SCOPE_AGENT);
        if (arrived == NWG - 1) {
            __hip_atomic_store(cnt, 0, __ATOMIC_RELAXED, __HIP_MEMORY_SCOPE_AGENT);
            __hip_atomic_fetch_add(gen, 1, __ATOMIC_RELEASE, __HIP_MEMORY_SCOPE_AGENT);
        } else {
            while (__hip_atomic_load(gen, __ATOMIC_ACQUIRE,
                                     __HIP_MEMORY_SCOPE_AGENT) == g) {
                __builtin_amdgcn_s_sleep(2);
            }
        }
    }
    __syncthreads();
    __threadfence();
}

__device__ __forceinline__ float sigmoidf_(float x) {
    return 1.0f / (1.0f + __expf(-x));
}

__global__ __launch_bounds__(WGSIZE, 1)
void elo_gru_scan(const float* __restrict__ emb_table,
                  const float* __restrict__ Wp,   const float* __restrict__ bp,
                  const float* __restrict__ W_ih, const float* __restrict__ b_ih,
                  const float* __restrict__ W_hh, const float* __restrict__ b_hh,
                  const float* __restrict__ ln_g, const float* __restrict__ ln_b,
                  const float* __restrict__ Wi1,  const float* __restrict__ bi1,
                  const float* __restrict__ Wi2,  const float* __restrict__ bi2,
                  const float* __restrict__ feats_a, const float* __restrict__ feats_b,
                  const int* __restrict__ team_a, const int* __restrict__ team_b,
                  const int* __restrict__ a_won,
                  float* __restrict__ out, float* __restrict__ ws) {
    extern __shared__ __align__(16) __bf16 dyn[];
    __bf16* WpS  = dyn;
    __bf16* WhhS = WpS  + WP_ELEMS;
    __bf16* WihS = WhhS + WHH_ELEMS;
    __bf16* Wi1S = WihS + WIH_ELEMS;

    __shared__ __align__(16) __bf16 combS[3 * CINP];   // rows 0,1 + zero row
    __shared__ __align__(16) __bf16 projS[3 * D_DIM];  // rows 0,1 + zero row
    __shared__ float  accS[8 * 32];       // <=8 tiles × (2 rows × 16 cols)
    __shared__ float  updB[2 * 2 * D_DIM];// post-LN rows, double buffered
    __shared__ float  redS[4];
    __shared__ float  impS[2];

    const int tid = threadIdx.x, wg = blockIdx.x;
    const int lane = tid & 31;
    const int wave = __builtin_amdgcn_readfirstlane(tid >> 5);  // wave-uniform

    int*   bar_cnt = (int*)ws;
    int*   bar_gen = ((int*)ws) + 1;
    float* ws_proj = ws + 64;                  // 2×512
    float* ws_gh   = ws_proj + 2 * D_DIM;      // 2×1536
    float* ws_gi   = ws_gh   + 2 * 3 * D_DIM;  // 2×1536
    float* ws_hid  = ws_gi   + 2 * 3 * D_DIM;  // 2×128
    float* embO = out;                         // working + final embedding
    float* ctx  = out + T_TEAMS * D_DIM;       // contexts (G,2,D)

    // ---- prologue: seed working emb, zero rows, weights -> LDS fragments ----
    for (int i = wg * WGSIZE + tid; i < T_TEAMS * D_DIM; i += NWG * WGSIZE)
        embO[i] = emb_table[i];
    for (int i = tid; i < CINP; i += WGSIZE)  combS[2 * CINP + i]  = (__bf16)0.0f;
    for (int i = tid; i < D_DIM; i += WGSIZE) projS[2 * D_DIM + i] = (__bf16)0.0f;
    load_frags(WpS,  Wp,   CIN_,  wg * 32, 2, KT_P, tid);
    load_frags(WhhS, W_hh, D_DIM, wg * 96, 6, KT_D, tid);
    load_frags(WihS, W_ih, D_DIM, wg * 96, 6, KT_D, tid);
    if (wg < 8) load_frags(Wi1S, Wi1, D_DIM, wg * 16, 1, KT_D, tid);
    grid_barrier(bar_cnt, bar_gen);

    int p_ta = -1, p_tb = -1;   // teams updated at the previous step
    for (int g = 0; g < G_GAMES; ++g) {
        const int ta = team_a[g], tb = team_b[g];
        const float aw = (float)a_won[g];
        const float* updPrev = updB + ((g & 1) ^ 1) * (2 * D_DIM);
        float*       updCur  = updB + (g & 1) * (2 * D_DIM);
        if (g + 1 < G_GAMES) {  // global_prefetch_b8 next game's features
            __builtin_prefetch(&feats_a[(size_t)(g + 1) * F_DIM], 0, 1);
            __builtin_prefetch(&feats_b[(size_t)(g + 1) * F_DIM], 0, 1);
        }

        // current embedding row read with 1-step local forwarding
        // (check p_tb first: .at[tb] wins when ta==tb in the reference)
        auto eread = [&](int row, int d) -> float {
            if (row == p_tb) return updPrev[D_DIM + d];
            if (row == p_ta) return updPrev[d];
            return embO[row * D_DIM + d];
        };

        // ---- build comb rows 0,1 (2×CINP) in LDS ----
        for (int idx = tid; idx < 2 * CINP; idx += WGSIZE) {
            int m = idx / CINP, k = idx - m * CINP;
            int self = m ? tb : ta, opp = m ? ta : tb;
            float v;
            if (k < 512)        v = eread(self, k);
            else if (k < 1024)  v = eread(opp, k - 512);
            else if (k < 1052)  v = (m ? feats_b : feats_a)[(size_t)g * F_DIM + (k - 1024)];
            else if (k == 1052) v = m ? (1.0f - aw) : aw;
            else                v = 0.0f;
            combS[idx] = (__bf16)v;
        }
        accS[tid] = 0.0f;
        __syncthreads();

        // ---- phase A: proj tiles (slots 0..1) + gh tiles (slots 2..7) ----
        {
            v8f accP[2] = {};
            v8f accH[6] = {};
            for (int kt = wave; kt < KT_P; kt += NWAVES) {
                v16bf a = build_afrag(combS, CINP, kt * 32, lane);
#pragma unroll
                for (int t = 0; t < 2; ++t) {
                    v16bf b = *(const v16bf*)&WpS[(size_t)((t * KT_P + kt) * 32 + lane) * 16];
                    accP[t] = __builtin_amdgcn_wmma_f32_16x16x32_bf16(
                        false, a, false, b, (short)0, accP[t], false, false);
                }
                if (kt < KT_D) {  // comb k<512 == e_self; reuse the A-fragment
#pragma unroll
                    for (int t = 0; t < 6; ++t) {
                        v16bf b = *(const v16bf*)&WhhS[(size_t)((t * KT_D + kt) * 32 + lane) * 16];
                        accH[t] = __builtin_amdgcn_wmma_f32_16x16x32_bf16(
                            false, a, false, b, (short)0, accH[t], false, false);
                    }
                }
            }
            if (lane < 16) {  // rows M=0,1 live in c[0],c[1] on lanes 0..15
#pragma unroll
                for (int t = 0; t < 2; ++t) {
                    atomicAdd(&accS[t * 32 + lane],      accP[t][0]);
                    atomicAdd(&accS[t * 32 + 16 + lane], accP[t][1]);
                }
#pragma unroll
                for (int t = 0; t < 6; ++t) {
                    atomicAdd(&accS[(2 + t) * 32 + lane],      accH[t][0]);
                    atomicAdd(&accS[(2 + t) * 32 + 16 + lane], accH[t][1]);
                }
            }
        }
        __syncthreads();
        if (tid < 64) {  // proj writeout: bias + relu -> ws + contexts
            int t = tid >> 5, m = (tid >> 4) & 1, n = tid & 15;
            int gcol = wg * 32 + t * 16 + n;
            float v = accS[t * 32 + m * 16 + n] + bp[gcol];
            v = v > 0.0f ? v : 0.0f;
            ws_proj[m * D_DIM + gcol] = v;
            ctx[((size_t)g * 2 + m) * D_DIM + gcol] = v;
        } else {         // gh writeout (192 entries)
            int i = tid - 64;
            int t = i >> 5, m = (i >> 4) & 1, n = i & 15;
            int gcol = wg * 96 + t * 16 + n;
            ws_gh[m * 3 * D_DIM + gcol] = accS[(2 + t) * 32 + m * 16 + n] + b_hh[gcol];
        }
        grid_barrier(bar_cnt, bar_gen);  // barrier 1: proj/gh published

        // ---- phase B: gi tiles (slots 0..5) + importance tile (slot 6) ----
        for (int idx = tid; idx < 2 * D_DIM; idx += WGSIZE)
            projS[idx] = (__bf16)ws_proj[idx];
        accS[tid] = 0.0f;
        __syncthreads();
        {
            v8f accI[6] = {};
            v8f accM = {};
            for (int kt = wave; kt < KT_D; kt += NWAVES) {
                v16bf a = build_afrag(projS, D_DIM, kt * 32, lane);
#pragma unroll
                for (int t = 0; t < 6; ++t) {
                    v16bf b = *(const v16bf*)&WihS[(size_t)((t * KT_D + kt) * 32 + lane) * 16];
                    accI[t] = __builtin_amdgcn_wmma_f32_16x16x32_bf16(
                        false, a, false, b, (short)0, accI[t], false, false);
                }
                if (wg < 8) {
                    v16bf b = *(const v16bf*)&Wi1S[(size_t)(kt * 32 + lane) * 16];
                    accM = __builtin_amdgcn_wmma_f32_16x16x32_bf16(
                        false, a, false, b, (short)0, accM, false, false);
                }
            }
            if (lane < 16) {
#pragma unroll
                for (int t = 0; t < 6; ++t) {
                    atomicAdd(&accS[t * 32 + lane],      accI[t][0]);
                    atomicAdd(&accS[t * 32 + 16 + lane], accI[t][1]);
                }
                if (wg < 8) {
                    atomicAdd(&accS[6 * 32 + lane],      accM[0]);
                    atomicAdd(&accS[6 * 32 + 16 + lane], accM[1]);
                }
            }
        }
        __syncthreads();
        if (tid < 192) {  // gi writeout
            int t = tid >> 5, m = (tid >> 4) & 1, n = tid & 15;
            int gcol = wg * 96 + t * 16 + n;
            ws_gi[m * 3 * D_DIM + gcol] = accS[t * 32 + m * 16 + n] + b_ih[gcol];
        } else if (wg < 8 && tid < 224) {  // hidden relu writeout
            int i = tid - 192;
            int m = i >> 4, n = i & 15;
            int gcol = wg * 16 + n;
            float v = accS[6 * 32 + m * 16 + n] + bi1[gcol];
            ws_hid[m * H_DIM + gcol] = v > 0.0f ? v : 0.0f;
        }
        grid_barrier(bar_cnt, bar_gen);  // barrier 2: gi/gh/hid published

        // ---- finale (ALL WGs, redundantly): importance, GRU, layernorm ----
        if (tid < 2) {
            float s = bi2[0];
            for (int h = 0; h < H_DIM; ++h)
                s += ws_hid[tid * H_DIM + h] * Wi2[h];
            impS[tid] = sigmoidf_(s);
        }
        if (tid < 4) redS[tid] = 0.0f;
        __syncthreads();
        for (int idx = tid; idx < 2 * D_DIM; idx += WGSIZE) {
            int m = idx >> 9, d = idx & 511;
            int row = m ? tb : ta;
            float ir = ws_gi[m * 1536 + d];
            float iz = ws_gi[m * 1536 + 512 + d];
            float in_ = ws_gi[m * 1536 + 1024 + d];
            float hr = ws_gh[m * 1536 + d];
            float hz = ws_gh[m * 1536 + 512 + d];
            float hn = ws_gh[m * 1536 + 1024 + d];
            float r = sigmoidf_(ir + hr);
            float z = sigmoidf_(iz + hz);
            float n = tanhf(in_ + r * hn);
            float e = eread(row, d);
            float nh = (1.0f - z) * n + z * e;
            float u = e + impS[m] * (nh - e);
            updCur[idx] = u;
            atomicAdd(&redS[m * 2 + 0], u);
            atomicAdd(&redS[m * 2 + 1], u * u);
        }
        __syncthreads();
        for (int idx = tid; idx < 2 * D_DIM; idx += WGSIZE) {
            int m = idx >> 9, d = idx & 511;
            int row = m ? tb : ta;
            float mu  = redS[m * 2 + 0] * (1.0f / 512.0f);
            float var = redS[m * 2 + 1] * (1.0f / 512.0f) - mu * mu;
            float y = (updCur[idx] - mu) * rsqrtf(var + 1e-5f) * ln_g[d] + ln_b[d];
            updCur[idx] = y;                            // local forward buffer
            if (wg == 0 && !(m == 0 && ta == tb))       // .at[tb] wins if equal
                embO[row * D_DIM + d] = y;              // consumed >=2 barrier
        }                                               // pairs later only
        __syncthreads();   // updCur complete before next step's comb build
        p_ta = ta; p_tb = tb;
    }
}

extern "C" void kernel_launch(void* const* d_in, const int* in_sizes, int n_in,
                              void* d_out, int out_size, void* d_ws, size_t ws_size,
                              hipStream_t stream) {
    (void)in_sizes; (void)n_in; (void)out_size; (void)ws_size;
    hipMemsetAsync(d_ws, 0, 256, stream);  // barrier counters
    elo_gru_scan<<<NWG, WGSIZE, DYN_BYTES, stream>>>(
        (const float*)d_in[0],  (const float*)d_in[1],  (const float*)d_in[2],
        (const float*)d_in[3],  (const float*)d_in[4],  (const float*)d_in[5],
        (const float*)d_in[6],  (const float*)d_in[7],  (const float*)d_in[8],
        (const float*)d_in[9],  (const float*)d_in[10], (const float*)d_in[11],
        (const float*)d_in[12], (const float*)d_in[13], (const float*)d_in[14],
        (const int*)d_in[15],   (const int*)d_in[16],   (const int*)d_in[17],
        (float*)d_out, (float*)d_ws);
}
